// SimpleRNN_5231270167134
// MI455X (gfx1250) — matrix operand order repaired and checked by reading, not compiled, not run on hardware
//
#include <hip/hip_runtime.h>

typedef _Float16 h16 __attribute__((ext_vector_type(16)));
typedef _Float16 h8  __attribute__((ext_vector_type(8)));
typedef float    f8  __attribute__((ext_vector_type(8)));
typedef float    f4  __attribute__((ext_vector_type(4)));

namespace {
constexpr int kT = 1000, kB = 64, kIn = 512, kH = 1024, kO = 64;
constexpr int kOPad = 80, kOCols = 65, kESize = 819;
constexpr int kNumWG = 16, kWaves = 8, kThreads = kWaves * 32;

// workspace byte offsets (total < 3.5 MB)
constexpr size_t WXT_B  = 0x00000000;  // eWx^T  [512][1024] f16 (1 MB)
constexpr size_t WHT_B  = 0x00100000;  // eWh^T  [1024][1024] f16 (2 MB)
constexpr size_t WOT_B  = 0x00300000;  // eW[o|v]^T padded [1024][80] f16 (160 KB)
constexpr size_t BPAD_B = 0x00330000;  // bias [80] f32
constexpr size_t CNT_B  = 0x00331000;  // u32 barrier counter
constexpr size_t OUT0_B = 0x00340000;  // hidden act ping [64][1024] f16 (128 KB)
constexpr size_t OUT1_B = 0x00360000;  // hidden act pong

constexpr int kSStride = 72;  // LDS row stride in halves (64 + 8 pad -> 2-way banks max)
constexpr size_t kSmemBytes = (size_t)(kIn + kH) * kSStride * sizeof(_Float16); // 221184 B
} // namespace

__device__ __forceinline__ h16 cat16(h8 lo, h8 hi) {
  return __builtin_shufflevector(lo, hi, 0,1,2,3,4,5,6,7,8,9,10,11,12,13,14,15);
}

__device__ __forceinline__ f8 wmma_f16(h16 a, h16 b, f8 c) {
  // D = A(16x32 f16) x B(32x16 f16) + C(16x16 f32)
  return __builtin_amdgcn_wmma_f32_16x16x32_f16(false, a, false, b, (short)0, c,
                                                false, false);
}

// ---------------------------------------------------------------------------
// Prep: build effective (relu+Dale-mask) weights transposed to [K][N] f16,
// padded readout weights, bias vector, zero ping-pong buffers, reset counter.
// ---------------------------------------------------------------------------
__global__ void rnn_prep(const float* __restrict__ Wx, const float* __restrict__ Wh,
                         const float* __restrict__ Wo, const float* __restrict__ Wv,
                         const float* __restrict__ bo, const float* __restrict__ bv,
                         unsigned char* __restrict__ ws) {
  const unsigned idx = blockIdx.x * blockDim.x + threadIdx.x;
  _Float16* wxt = (_Float16*)(ws + WXT_B);
  _Float16* wht = (_Float16*)(ws + WHT_B);
  _Float16* wot = (_Float16*)(ws + WOT_B);
  float* bpad   = (float*)(ws + BPAD_B);
  unsigned* cnt = (unsigned*)(ws + CNT_B);
  _Float16* ob0 = (_Float16*)(ws + OUT0_B);
  _Float16* ob1 = (_Float16*)(ws + OUT1_B);

  if (idx < (unsigned)(kIn * kH)) {            // eWx^T[k][n] = relu(Wx[n][k])
    unsigned k = idx >> 10, n = idx & (kH - 1);
    wxt[idx] = (_Float16)fmaxf(Wx[n * kIn + k], 0.f);
  }
  if (idx < (unsigned)(kH * kH)) {             // eWh^T[k][n] = relu(Wh[n][k])*sign_k*(n!=k)
    unsigned k = idx >> 10, n = idx & (kH - 1);
    float v = (n == k) ? 0.f : fmaxf(Wh[n * kH + k], 0.f);
    wht[idx] = (_Float16)((k < (unsigned)kESize) ? v : -v);
  }
  if (idx < (unsigned)(kH * kOPad)) {          // [eWo ; eWv ; 0-pad]^T, col-masked
    unsigned k = idx / kOPad, n = idx % kOPad;
    float v = 0.f;
    if (k < (unsigned)kESize) {
      if (n < (unsigned)kO) v = fmaxf(Wo[n * kH + k], 0.f);
      else if (n == (unsigned)kO) v = fmaxf(Wv[k], 0.f);
    }
    wot[idx] = (_Float16)v;
  }
  if (idx < (unsigned)kOPad)
    bpad[idx] = (idx < (unsigned)kO) ? bo[idx] : ((idx == (unsigned)kO) ? bv[0] : 0.f);
  if (idx < (unsigned)(kB * kH)) { ob0[idx] = (_Float16)0.f; ob1[idx] = (_Float16)0.f; }
  if (idx == 0) *cnt = 0u;
}

// ---------------------------------------------------------------------------
// Persistent RNN: 16 WGs (one 64-col slice of H each) x 8 waves x 1000 steps.
// State lives in f32 VGPRs; activations ping-pong through L2; weights in LDS.
// ---------------------------------------------------------------------------
__global__ __launch_bounds__(kThreads, 1)
void rnn_persist(const float* __restrict__ x, const float* __restrict__ bh,
                 unsigned char* __restrict__ ws, float* __restrict__ yout) {
  extern __shared__ char smem[];
  _Float16* sWx = (_Float16*)smem;                       // [512][72]
  _Float16* sWh = sWx + (size_t)kIn * kSStride;          // [1024][72]

  const int tid  = threadIdx.x;
  const int lane = tid & 31;
  const int w    = tid >> 5;
  const int wg   = blockIdx.x;
  const int mrow  = lane & 15;          // tile row (A) / tile col (B,C,D)
  const int khalf = (lane >> 4) * 8;    // A-frag K sub-offset per ISA layout
  const int m0    = (w & 3) * 16;       // batch-tile base row
  const int ng    = w >> 2;             // n-pair group (0 or 1)
  const int lc0   = ng * 32;            // local col base within 64-col slice
  const int gc0   = wg * 64 + lc0;      // global col base

  const _Float16* wxt = (const _Float16*)(ws + WXT_B);
  const _Float16* wht = (const _Float16*)(ws + WHT_B);
  const _Float16* wot = (const _Float16*)(ws + WOT_B);
  const float* bpad   = (const float*)(ws + BPAD_B);
  unsigned* cnt       = (unsigned*)(ws + CNT_B);
  _Float16* ob0       = (_Float16*)(ws + OUT0_B);
  _Float16* ob1       = (_Float16*)(ws + OUT1_B);

  // Stage this WG's weight slices into LDS once (row-padded).
  for (int i = tid; i < kIn * 8; i += kThreads) {
    int r = i >> 3, c = (i & 7) * 8;
    *(h8*)(sWx + (size_t)r * kSStride + c) = *(const h8*)(wxt + (size_t)r * kH + wg * 64 + c);
  }
  for (int i = tid; i < kH * 8; i += kThreads) {
    int r = i >> 3, c = (i & 7) * 8;
    *(h8*)(sWh + (size_t)r * kSStride + c) = *(const h8*)(wht + (size_t)r * kH + wg * 64 + c);
  }
  __syncthreads();

  const int gw  = wg * kWaves + w;      // global wave id: first 20 own a readout tile
  const int mty = gw & 3, nty = gw >> 2;

  f8 st0 = {}, st1 = {};                // persistent f32 state for the 2 owned tiles
  const float bh0 = bh[gc0 + mrow];
  const float bh1 = bh[gc0 + 16 + mrow];

#pragma unroll 1
  for (int t = 0; t < kT; ++t) {
    const int p = t & 1;
    const _Float16* oprev = p ? ob1 : ob0;
    _Float16* onext       = p ? ob0 : ob1;
    f8 acc0 = {}, acc1 = {};

    // ---- input GEMM: x_t[64x512] (f32 -> f16 on the fly) ----
    const float* xrow = x + ((size_t)t * kB + (m0 + mrow)) * kIn + khalf;
    if (t + 1 < kT) __builtin_prefetch((const void*)(xrow + (size_t)kB * kIn), 0, 1);
#pragma unroll 2
    for (int ks = 0; ks < kIn / 32; ++ks) {
      const int k0 = ks * 32;
      f4 c0 = *(const f4*)(xrow + k0);
      f4 c1 = *(const f4*)(xrow + k0 + 4);
      f4 c2 = *(const f4*)(xrow + k0 + 16);
      f4 c3 = *(const f4*)(xrow + k0 + 20);
      h16 a;
#pragma unroll
      for (int i = 0; i < 4; ++i) {
        a[i]      = (_Float16)c0[i];  a[4 + i]  = (_Float16)c1[i];
        a[8 + i]  = (_Float16)c2[i];  a[12 + i] = (_Float16)c3[i];
      }
      const _Float16* bp = sWx + (size_t)(k0 + khalf) * kSStride + lc0 + mrow;
      h16 b0, b1;
#pragma unroll
      for (int s = 0; s < 16; ++s) {
        const int kk = (s < 8) ? s : s + 8;
        b0[s] = bp[(size_t)kk * kSStride];
        b1[s] = bp[(size_t)kk * kSStride + 16];
      }
      acc0 = wmma_f16(a, b0, acc0);
      acc1 = wmma_f16(a, b1, acc1);
    }
    // ---- recurrent GEMM: out[64x1024] (f16 from L2 ping buffer) ----
    const _Float16* orow = oprev + (size_t)(m0 + mrow) * kH + khalf;
#pragma unroll 2
    for (int ks = 0; ks < kH / 32; ++ks) {
      const int k0 = ks * 32;
      h16 a = cat16(*(const h8*)(orow + k0), *(const h8*)(orow + k0 + 16));
      const _Float16* bp = sWh + (size_t)(k0 + khalf) * kSStride + lc0 + mrow;
      h16 b0, b1;
#pragma unroll
      for (int s = 0; s < 16; ++s) {
        const int kk = (s < 8) ? s : s + 8;
        b0[s] = bp[(size_t)kk * kSStride];
        b1[s] = bp[(size_t)kk * kSStride + 16];
      }
      acc0 = wmma_f16(a, b0, acc0);
      acc1 = wmma_f16(a, b1, acc1);
    }

    // ---- leaky Euler + retanh, publish activations ----
    {
      const int rbase = m0 + (lane >> 4) * 8;
#pragma unroll
      for (int g = 0; g < 8; ++g) {
        float s = 0.8f * st0[g] + 0.2f * (acc0[g] + bh0);
        st0[g] = s;
        onext[(size_t)(rbase + g) * kH + gc0 + mrow] = (_Float16)tanhf(fmaxf(s, 0.f));
        s = 0.8f * st1[g] + 0.2f * (acc1[g] + bh1);
        st1[g] = s;
        onext[(size_t)(rbase + g) * kH + gc0 + 16 + mrow] = (_Float16)tanhf(fmaxf(s, 0.f));
      }
    }

    // ---- device-wide step barrier (atomic counter + cluster barrier path) ----
    __builtin_amdgcn_fence(__ATOMIC_RELEASE, "agent");
    __syncthreads();
    if (tid == 0) {
      __hip_atomic_fetch_add(cnt, 1u, __ATOMIC_RELEASE, __HIP_MEMORY_SCOPE_AGENT);
      const unsigned target = (unsigned)kNumWG * (unsigned)(t + 1);
      while (__hip_atomic_load(cnt, __ATOMIC_ACQUIRE, __HIP_MEMORY_SCOPE_AGENT) < target)
        __builtin_amdgcn_s_sleep(1);
    }
    __syncthreads();
    __builtin_amdgcn_s_cluster_barrier();  // NOP if not dispatched as a cluster
    __builtin_amdgcn_fence(__ATOMIC_ACQUIRE, "agent");

    // ---- readout head y_t = out' @ [eWo;eWv]^T + b  (20 tiles on waves 0..19) ----
    if (gw < 20) {
      f8 ya = {};
      const _Float16* yrow = onext + (size_t)(mty * 16 + mrow) * kH + khalf;
#pragma unroll 2
      for (int ks = 0; ks < kH / 32; ++ks) {
        const int k0 = ks * 32;
        h16 a = cat16(*(const h8*)(yrow + k0), *(const h8*)(yrow + k0 + 16));
        const _Float16* bp = wot + (size_t)(k0 + khalf) * kOPad + nty * 16 + mrow;
        h16 b;
#pragma unroll
        for (int s = 0; s < 16; ++s) b[s] = bp[(size_t)((s < 8) ? s : s + 8) * kOPad];
        ya = wmma_f16(a, b, ya);
      }
      const int c = nty * 16 + mrow;
      if (c < kOCols) {
        const float bb = bpad[c];
        float* yo = yout + (size_t)t * (kB * kOCols) +
                    (size_t)(mty * 16 + (lane >> 4) * 8) * kOCols + c;
#pragma unroll
        for (int g = 0; g < 8; ++g) yo[(size_t)g * kOCols] = ya[g] + bb;
      }
    }
  }
}

extern "C" void kernel_launch(void* const* d_in, const int* in_sizes, int n_in,
                              void* d_out, int out_size, void* d_ws, size_t ws_size,
                              hipStream_t stream) {
  (void)in_sizes; (void)n_in; (void)out_size; (void)ws_size;
  const float* x  = (const float*)d_in[0];
  const float* Wx = (const float*)d_in[1];
  const float* Wh = (const float*)d_in[2];
  const float* bh = (const float*)d_in[3];
  const float* Wo = (const float*)d_in[4];
  const float* bo = (const float*)d_in[5];
  const float* Wv = (const float*)d_in[6];
  const float* bv = (const float*)d_in[7];
  unsigned char* ws = (unsigned char*)d_ws;
  float* yout = (float*)d_out;

  rnn_prep<<<(kH * kH) / 256, 256, 0, stream>>>(Wx, Wh, Wo, Wv, bo, bv, ws);
  rnn_persist<<<kNumWG, kThreads, kSmemBytes, stream>>>(x, bh, ws, yout);
}
